// DCRNNModel_nextTimePred_57354993271298
// MI455X (gfx1250) — compile-verified, matrix operand order checked by reading, not run on hardware
//
#include <hip/hip_runtime.h>

// ---------------------------------------------------------------------------
// DCRNN (diffusion-conv GRU seq2seq) for MI455X / gfx1250, wave32 + WMMA.
//
// GEMMs run v_wmma_f32_16x16x32_bf16 (f32 accumulate) in a *transposed*
// formulation so A, B, C accesses are all 128-bit contiguous:
//   diffusion:  C'(BF x N) = XS^T (BF x N) @ S       (row-major S IS the
//               column-contiguous B operand of the transposed product)
//   projection: C (NB x O) = Xcat (NB x Kpad) @ W^T  (W stored transposed)
// A tiles (32x32) are DMA'd to double-buffered LDS with
// global_load_async_to_lds_b128, software-pipelined one tile ahead
// (s_wait_asynccnt 0x1) so the DMA overlaps the WMMAs. Each wave holds two
// row-tile accumulators sharing one B fragment (2 WMMAs per B load).
// ---------------------------------------------------------------------------

typedef __attribute__((ext_vector_type(16))) __bf16 v16bf;
typedef __attribute__((ext_vector_type(8)))  __bf16 v8bf;
typedef __attribute__((ext_vector_type(8)))  float  v8f;

#define NN   256   // nodes
#define HID  64
#define BB   64    // batch
#define TT   12
#define MM   5     // identity + 2 hops * 2 supports

// 16x32 bf16 fragment: element i has K = ((i>>3)<<4) + half*8 + (i&7), i.e.
// two contiguous 8-element runs -> two 128-bit loads.
__device__ __forceinline__ v16bf load_frag_row(const __bf16* __restrict__ row, int half) {
  v8bf lo = *(const v8bf*)(row + half * 8);
  v8bf hi = *(const v8bf*)(row + 16 + half * 8);
  return __builtin_shufflevector(lo, hi, 0, 1, 2, 3, 4, 5, 6, 7,
                                 8, 9, 10, 11, 12, 13, 14, 15);
}

// Async DMA of one 16-byte chunk of a 32x32 bf16 A tile into LDS.
__device__ __forceinline__ void async_fill_tile(const __bf16* __restrict__ Ag, long lda,
                                                int rowBase, int kko,
                                                __bf16* __restrict__ dst) {
  int fr = threadIdx.x >> 2;            // 0..31 tile row
  int fc = (threadIdx.x & 3) * 8;       // 0,8,16,24 (8 bf16 = 16B)
  const __bf16* gsrc = Ag + (long)(rowBase + fr) * lda + kko + fc;
  unsigned lofs = (unsigned)(size_t)&dst[fr * 32 + fc];
  asm volatile("global_load_async_to_lds_b128 %0, %1, off"
               :: "v"(lofs), "v"((unsigned long long)(size_t)gsrc) : "memory");
  __builtin_prefetch(gsrc + 64, 0, 1);  // speculative: tile after next
}

// ------------------------- conversion / init kernels -----------------------

__global__ void f32_to_bf16_k(const float* __restrict__ src, __bf16* __restrict__ dst, int n) {
  int i = blockIdx.x * blockDim.x + threadIdx.x;
  if (i < n) dst[i] = (__bf16)src[i];
}

// W (K x O) f32 -> Wp^T (O x Kpad) bf16, zero padded in K.
__global__ void pack_weight_k(const float* __restrict__ W, __bf16* __restrict__ WpT,
                              int K, int Kpad, int O) {
  int i = blockIdx.x * blockDim.x + threadIdx.x;
  if (i >= Kpad * O) return;
  int o = i / Kpad, k = i % Kpad;
  WpT[i] = (k < K) ? (__bf16)W[k * O + o] : (__bf16)0.0f;
}

__global__ void zero_f32_k(float* __restrict__ p, int n) {
  int i = blockIdx.x * blockDim.x + threadIdx.x;
  if (i < n) p[i] = 0.0f;
}

// ------------------------------- pack XS -----------------------------------
// Builds XS^T (BF x N) in f32 + bf16 for the diffusion GEMMs, and writes the
// identity hop (m=0) plus zero K-padding into packed Xcat (rows = n*B+b,
// cols = f*M+m, bf16) consumed by the projection GEMM.
__global__ void pack_xs_k(const float* __restrict__ xsrc, long xoff, long xbstride, int in_dim,
                          const float* __restrict__ h, const float* __restrict__ rgate,
                          __bf16* __restrict__ XSb, float* __restrict__ XSf,
                          __bf16* __restrict__ Xcat, int F, int Kpad) {
  long idx = (long)blockIdx.x * blockDim.x + threadIdx.x;
  long total = (long)NN * BB * Kpad;
  if (idx >= total) return;
  int k = (int)(idx % Kpad);
  long nb = idx / Kpad;
  int b = (int)(nb % BB);
  int n = (int)(nb / BB);
  if (k >= F * MM) { Xcat[idx] = (__bf16)0.0f; return; }
  int f = k / MM, m = k % MM;
  if (m != 0) return;                       // hops 1..4 written by GEMM epilogues
  float v;
  if (f < in_dim) {
    v = xsrc[xoff + (long)b * xbstride + (long)n * in_dim + f];
  } else {
    int j = f - in_dim;
    v = h[((long)b * NN + n) * HID + j];
    if (rgate) v *= rgate[((long)n * BB + b) * (2 * HID) + j];   // r * h
  }
  Xcat[idx] = (__bf16)v;
  long xsi = ((long)b * F + f) * NN + n;    // XS^T: row c = b*F+f, col n
  XSf[xsi] = v;
  XSb[xsi] = (__bf16)v;
}

// -------------------------- diffusion GEMM (WMMA) --------------------------
// C'(BF x 256) = Ab(BF x 256) @ S.  Block = 32 rows x 64 cols, wave = one
// col tile x two row tiles (shared B fragment, two accumulators).
// mode 1: x1 = S@xs  -> write T1f/T1b (BF x 256) and Xcat column m=mcol
// mode 2: x2 = 2*S@x1 - xs (Chebyshev) -> write Xcat column m=mcol
__global__ __launch_bounds__(128)
void diff_gemm_k(const __bf16* __restrict__ Ab, const __bf16* __restrict__ Sb,
                 const float* __restrict__ X0f, float* __restrict__ T1f,
                 __bf16* __restrict__ T1b, __bf16* __restrict__ Xcat,
                 int F, int Kpad, int mcol, int mode) {
  __shared__ __align__(16) __bf16 sA[2][32 * 32];
  const int wave = threadIdx.x >> 5;
  const int lane = threadIdx.x & 31;
  const int half = lane >> 4;
  const int l15  = lane & 15;
  const int rowBase = blockIdx.y * 32;                 // activation rows c = b*F+f
  const int colBase = blockIdx.x * 64 + wave * 16;     // node columns
  const __bf16* bcol = Sb + (long)(colBase + l15) * NN;

  v8f acc0 = {}, acc1 = {};
  async_fill_tile(Ab, NN, rowBase, 0, sA[0]);          // prologue: tile 0
  for (int kk = 0; kk < NN; kk += 32) {
    const int cur = (kk >> 5) & 1;
    __syncthreads();                                   // buf[cur^1] readers done
    if (kk + 32 < NN) {
      async_fill_tile(Ab, NN, rowBase, kk + 32, sA[cur ^ 1]);
      asm volatile("s_wait_asynccnt 0x1" ::: "memory");  // cur landed, next in flight
    } else {
      asm volatile("s_wait_asynccnt 0x0" ::: "memory");
    }
    __syncthreads();
    v16bf a0 = load_frag_row(&sA[cur][l15 * 32], half);         // rows r..r+15
    v16bf a1 = load_frag_row(&sA[cur][(l15 + 16) * 32], half);  // rows r+16..r+31
    v16bf b  = load_frag_row(bcol + kk, half);                  // shared B frag
    acc0 = __builtin_amdgcn_wmma_f32_16x16x32_bf16(false, a0, false, b,
                                                   (short)0, acc0, false, false);
    acc1 = __builtin_amdgcn_wmma_f32_16x16x32_bf16(false, a1, false, b,
                                                   (short)0, acc1, false, false);
  }
#pragma unroll
  for (int rt = 0; rt < 2; ++rt) {
    v8f acc = rt ? acc1 : acc0;
#pragma unroll
    for (int e = 0; e < 8; ++e) {
      int c = rowBase + rt * 16 + half * 8 + e;  // C layout: vgpr e -> M = e+8*half
      int m = colBase + l15;                     // output node
      float v = acc[e];
      int bi = c / F, f = c % F;
      if (mode == 1) {
        T1f[(long)c * NN + m] = v;
        T1b[(long)c * NN + m] = (__bf16)v;
      } else {
        v = 2.0f * v - X0f[(long)c * NN + m];
      }
      Xcat[((long)(m * BB + bi)) * Kpad + f * MM + mcol] = (__bf16)v;
    }
  }
}

// ------------------------- projection GEMM (WMMA) --------------------------
// out(16384 x O) = Xcat(16384 x Kpad) @ WpT^T + bias.  Same 32x64 blocking.
// act==1: sigmoid (gates r,u), act==2: tanh (candidate).
__global__ __launch_bounds__(128)
void proj_gemm_k(const __bf16* __restrict__ A, const __bf16* __restrict__ WpT,
                 const float* __restrict__ bias, float* __restrict__ out,
                 int Kpad, int O, int act) {
  __shared__ __align__(16) __bf16 sA[2][32 * 32];
  const int wave = threadIdx.x >> 5;
  const int lane = threadIdx.x & 31;
  const int half = lane >> 4;
  const int l15  = lane & 15;
  const int rowBase = blockIdx.y * 32;
  const int colBase = blockIdx.x * 64 + wave * 16;
  const __bf16* bcol = WpT + (long)(colBase + l15) * Kpad;

  v8f acc0 = {}, acc1 = {};
  async_fill_tile(A, Kpad, rowBase, 0, sA[0]);
  for (int kk = 0; kk < Kpad; kk += 32) {
    const int cur = (kk >> 5) & 1;
    __syncthreads();
    if (kk + 32 < Kpad) {
      async_fill_tile(A, Kpad, rowBase, kk + 32, sA[cur ^ 1]);
      asm volatile("s_wait_asynccnt 0x1" ::: "memory");
    } else {
      asm volatile("s_wait_asynccnt 0x0" ::: "memory");
    }
    __syncthreads();
    v16bf a0 = load_frag_row(&sA[cur][l15 * 32], half);
    v16bf a1 = load_frag_row(&sA[cur][(l15 + 16) * 32], half);
    v16bf b  = load_frag_row(bcol + kk, half);
    acc0 = __builtin_amdgcn_wmma_f32_16x16x32_bf16(false, a0, false, b,
                                                   (short)0, acc0, false, false);
    acc1 = __builtin_amdgcn_wmma_f32_16x16x32_bf16(false, a1, false, b,
                                                   (short)0, acc1, false, false);
  }
#pragma unroll
  for (int rt = 0; rt < 2; ++rt) {
    v8f acc = rt ? acc1 : acc0;
#pragma unroll
    for (int e = 0; e < 8; ++e) {
      int r = rowBase + rt * 16 + half * 8 + e;
      int o = colBase + l15;
      float v = acc[e] + bias[o];
      v = (act == 1) ? (1.0f / (1.0f + __expf(-v))) : tanhf(v);
      out[(long)r * O + o] = v;
    }
  }
}

// ------------------------------- GRU update --------------------------------
__global__ void gru_update_k(const float* __restrict__ ru, const float* __restrict__ cand,
                             float* __restrict__ h) {
  int idx = blockIdx.x * blockDim.x + threadIdx.x;   // B*N*H
  if (idx >= BB * NN * HID) return;
  int j = idx % HID;
  int n = (idx / HID) % NN;
  int b = idx / (HID * NN);
  float u = ru[((long)n * BB + b) * (2 * HID) + HID + j];
  float c = cand[((long)n * BB + b) * HID + j];
  float hv = h[idx];                                  // h layout b*N*H + n*H + j
  h[idx] = u * hv + (1.0f - u) * c;
}

// --------------------------- final projection ------------------------------
__global__ void out_proj_k(const float* __restrict__ h, const float* __restrict__ W,
                           const float* __restrict__ bias, float* __restrict__ dout,
                           float* __restrict__ po, int t) {
  int idx = blockIdx.x * blockDim.x + threadIdx.x;   // B*N
  if (idx >= BB * NN) return;
  int n = idx % NN, b = idx / NN;
  const float* hp = h + (long)idx * HID;
  float s = bias[0];
#pragma unroll 8
  for (int j = 0; j < HID; ++j) s += hp[j] * W[j];
  dout[((long)b * TT + t) * NN + n] = s;
  po[idx] = s;                                        // decoder feedback input
}

// ------------------------------- host side ---------------------------------

extern "C" void kernel_launch(void* const* d_in, const int* in_sizes, int n_in,
                              void* d_out, int out_size, void* d_ws, size_t ws_size,
                              hipStream_t stream) {
  (void)in_sizes; (void)n_in; (void)out_size; (void)ws_size;
  const float* enc_in = (const float*)d_in[0];
  const float* S0     = (const float*)d_in[2];
  const float* S1     = (const float*)d_in[3];
  struct Layer { const float *Wg, *bg, *Wc, *bc; int F; int Kpad; };
  Layer L[4] = {
    {(const float*)d_in[4],  (const float*)d_in[5],  (const float*)d_in[6],  (const float*)d_in[7],  HID + 2, 352},
    {(const float*)d_in[8],  (const float*)d_in[9],  (const float*)d_in[10], (const float*)d_in[11], 2 * HID, 640},
    {(const float*)d_in[12], (const float*)d_in[13], (const float*)d_in[14], (const float*)d_in[15], HID + 1, 352},
    {(const float*)d_in[16], (const float*)d_in[17], (const float*)d_in[18], (const float*)d_in[19], 2 * HID, 640},
  };
  const float* projW = (const float*)d_in[20];
  const float* projB = (const float*)d_in[21];

  char* w = (char*)d_ws;
  auto alloc = [&](size_t bytes) { char* p = w; w += (bytes + 255) & ~(size_t)255; return p; };
  __bf16* Sb[2];
  Sb[0] = (__bf16*)alloc((size_t)NN * NN * 2);
  Sb[1] = (__bf16*)alloc((size_t)NN * NN * 2);
  __bf16 *WgP[4], *WcP[4];
  for (int i = 0; i < 4; ++i) {
    WgP[i] = (__bf16*)alloc((size_t)L[i].Kpad * 2 * HID * 2);
    WcP[i] = (__bf16*)alloc((size_t)L[i].Kpad * HID * 2);
  }
  float*  XSf  = (float*) alloc((size_t)NN * BB * 128 * 4);   // XS^T (BF x N)
  __bf16* XSb  = (__bf16*)alloc((size_t)NN * BB * 128 * 2);
  float*  T1f  = (float*) alloc((size_t)NN * BB * 128 * 4);   // x1^T (BF x N)
  __bf16* T1b  = (__bf16*)alloc((size_t)NN * BB * 128 * 2);
  __bf16* Xcat = (__bf16*)alloc((size_t)NN * BB * 640 * 2);
  float*  ru   = (float*) alloc((size_t)NN * BB * 2 * HID * 4);
  float*  cand = (float*) alloc((size_t)NN * BB * HID * 4);
  float*  hA   = (float*) alloc((size_t)BB * NN * HID * 4);
  float*  hB   = (float*) alloc((size_t)BB * NN * HID * 4);
  float*  po   = (float*) alloc((size_t)BB * NN * 4);

  // one-time conversions + state init (re-done every call: deterministic)
  f32_to_bf16_k<<<(NN * NN + 255) / 256, 256, 0, stream>>>(S0, Sb[0], NN * NN);
  f32_to_bf16_k<<<(NN * NN + 255) / 256, 256, 0, stream>>>(S1, Sb[1], NN * NN);
  for (int i = 0; i < 4; ++i) {
    int ng = L[i].Kpad * 2 * HID, nc = L[i].Kpad * HID;
    pack_weight_k<<<(ng + 255) / 256, 256, 0, stream>>>(L[i].Wg, WgP[i], L[i].F * MM, L[i].Kpad, 2 * HID);
    pack_weight_k<<<(nc + 255) / 256, 256, 0, stream>>>(L[i].Wc, WcP[i], L[i].F * MM, L[i].Kpad, HID);
  }
  zero_f32_k<<<(BB * NN * HID + 255) / 256, 256, 0, stream>>>(hA, BB * NN * HID);
  zero_f32_k<<<(BB * NN * HID + 255) / 256, 256, 0, stream>>>(hB, BB * NN * HID);
  zero_f32_k<<<(BB * NN + 255) / 256, 256, 0, stream>>>(po, BB * NN);  // go symbol

  auto diff_conv = [&](const float* xsrc, long xoff, long xbstride, int in_dim,
                       const float* h, const float* rgate, int li,
                       const __bf16* WpT, const float* bias, int O, int act, float* outBuf) {
    int F = L[li].F, Kpad = L[li].Kpad;
    long total = (long)NN * BB * Kpad;
    pack_xs_k<<<(unsigned)((total + 255) / 256), 256, 0, stream>>>(
        xsrc, xoff, xbstride, in_dim, h, rgate, XSb, XSf, Xcat, F, Kpad);
    dim3 g(NN / 64, (BB * F) / 32);
    for (int s = 0; s < 2; ++s) {   // per support: x1 = S@xs ; x2 = 2*S@x1 - xs
      diff_gemm_k<<<g, 128, 0, stream>>>(XSb, Sb[s], nullptr, T1f, T1b, Xcat,
                                         F, Kpad, 1 + s * 2, 1);
      diff_gemm_k<<<g, 128, 0, stream>>>(T1b, Sb[s], XSf, nullptr, nullptr, Xcat,
                                         F, Kpad, 2 + s * 2, 2);
    }
    dim3 gp(O / 64, (NN * BB) / 32);
    proj_gemm_k<<<gp, 128, 0, stream>>>(Xcat, WpT, bias, outBuf, Kpad, O, act);
  };

  auto cell = [&](const float* xsrc, long xoff, long xbstride, int in_dim, int li, float* h) {
    diff_conv(xsrc, xoff, xbstride, in_dim, h, nullptr, li, WgP[li], L[li].bg, 2 * HID, 1, ru);
    diff_conv(xsrc, xoff, xbstride, in_dim, h, ru,      li, WcP[li], L[li].bc, HID,     2, cand);
    gru_update_k<<<(BB * NN * HID + 255) / 256, 256, 0, stream>>>(ru, cand, h);
  };

  // encoder (layers interleaved per timestep — identical to sequential scans)
  for (int t = 0; t < TT; ++t) {
    cell(enc_in, (long)t * NN * 2, (long)TT * NN * 2, 2, 0, hA);
    cell(hA, 0, (long)NN * HID, HID, 1, hB);
  }
  // decoder: go symbol = zeros, feedback = own projection
  for (int t = 0; t < TT; ++t) {
    cell(po, 0, NN, 1, 2, hA);
    cell(hA, 0, (long)NN * HID, HID, 3, hB);
    out_proj_k<<<(BB * NN + 255) / 256, 256, 0, stream>>>(hB, projW, projB,
                                                          (float*)d_out, po, t);
  }
}
